// S1Net_29540785062207
// MI455X (gfx1250) — compile-verified, matrix operand order checked by reading, not run on hardware
//
#include <hip/hip_runtime.h>
#include <math.h>

// ---------------- problem constants (match reference) ----------------
#define N_NEURON   500000
#define N_CONN     8
#define SIM_TIME   200
#define I_MAX_F    20.0f
#define FIRE_TH    0.5f
#define IZH_TH     30.0f
#define IZH_A      0.02f
#define IZH_B      0.2f
#define IZH_CRST  (-65.0f)
#define IZH_D      8.0f
#define MEM0      (-70.0f)
#define U0        (-14.0f)
#define DT         0.5f
#define PI_F       3.14159265358979323846f

typedef float    v2f  __attribute__((ext_vector_type(2)));
typedef float    v8f  __attribute__((ext_vector_type(8)));
typedef _Float16 v16h __attribute__((ext_vector_type(16)));

// ---------------------------------------------------------------------
// Kernel 1: 21-entry Izhikevich spike-count LUT (I = 0..20) + flag reset.
// input_r = round(input_n * 20) with input_n in [0,1) -> integer in {0..20},
// so the whole 200-step scan collapses to a table lookup per neuron.
__global__ void izh_lut_kernel(float* __restrict__ lut,
                               unsigned int* __restrict__ fired_flag) {
  const int t = threadIdx.x;
  if (t == 31) *fired_flag = 0u;          // reset reduction flag every call
  if (t <= 20) {
    const float I = (float)t;
    float mem = MEM0, u = U0, acc = 0.0f;
    #pragma unroll 4
    for (int s = 0; s < SIM_TIME; ++s) {
      mem = mem + DT * (0.04f * mem * mem + 5.0f * mem + 140.0f - u + I);
      u   = u + DT * IZH_A * (IZH_B * mem - u);
      const float spike = (mem >= IZH_TH) ? 1.0f : 0.0f;
      mem = spike * IZH_CRST + (1.0f - spike) * mem;
      u   = u + spike * IZH_D;
      acc += spike;
    }
    lut[t] = acc;
  }
}

// ---------------------------------------------------------------------
// Kernel 2: main streaming pass.
//   FR_W via WMMA (A = ones -> D rows = column sums of B = FR.*W),
//   then tanh/relu pipeline, LUT lookup, and wave-ballot fired-any OR.
__global__ __launch_bounds__(256) void neuron_kernel(
    const float* __restrict__ input, const float* __restrict__ FR,
    const float* __restrict__ C,     const float* __restrict__ W,
    const float* __restrict__ lut,   unsigned int* __restrict__ fired_flag,
    float* __restrict__ frn_out,     float* __restrict__ inpn_out) {

  const int lane  = threadIdx.x & 31;                       // wave32
  const int wbase = blockIdx.x * blockDim.x + (threadIdx.x & ~31);
  const int i     = wbase + lane;                           // this lane's neuron
  const int nm1   = N_NEURON - 1;

  float frw;  // = sum_k FR[k,i] * W[k,i]

#if __has_builtin(__builtin_amdgcn_wmma_f32_16x16x4_f32)
  {
    // f32 path: B is 4x16 (K x N). Lane l owns column n = l%16; lane halves
    // carry K rows {0,1} / {2,3}. A = ones(16x4) => D[m,n] = sum_k B[k,n],
    // replicated over rows m, so c[0] holds the answer in every lane.
    const int col  = lane & 15;
    const int kh   = lane >> 4;
    const int n_lo = min(wbase + col,      nm1);
    const int n_hi = min(wbase + 16 + col, nm1);
    const int k0 = 2 * kh, k1 = k0 + 1;
    const v2f a = {1.0f, 1.0f};
    v8f c_lo = {}, c_hi = {};
    v2f b;
    // lo tile (neurons wbase..wbase+15): k rows {0..3} then {4..7}
    b[0] = FR[k0 * N_NEURON + n_lo] * W[k0 * N_NEURON + n_lo];
    b[1] = FR[k1 * N_NEURON + n_lo] * W[k1 * N_NEURON + n_lo];
    c_lo = __builtin_amdgcn_wmma_f32_16x16x4_f32(false, a, false, b, (short)0, c_lo, false, false);
    b[0] = FR[(k0 + 4) * N_NEURON + n_lo] * W[(k0 + 4) * N_NEURON + n_lo];
    b[1] = FR[(k1 + 4) * N_NEURON + n_lo] * W[(k1 + 4) * N_NEURON + n_lo];
    c_lo = __builtin_amdgcn_wmma_f32_16x16x4_f32(false, a, false, b, (short)0, c_lo, false, false);
    // hi tile (neurons wbase+16..wbase+31)
    b[0] = FR[k0 * N_NEURON + n_hi] * W[k0 * N_NEURON + n_hi];
    b[1] = FR[k1 * N_NEURON + n_hi] * W[k1 * N_NEURON + n_hi];
    c_hi = __builtin_amdgcn_wmma_f32_16x16x4_f32(false, a, false, b, (short)0, c_hi, false, false);
    b[0] = FR[(k0 + 4) * N_NEURON + n_hi] * W[(k0 + 4) * N_NEURON + n_hi];
    b[1] = FR[(k1 + 4) * N_NEURON + n_hi] * W[(k1 + 4) * N_NEURON + n_hi];
    c_hi = __builtin_amdgcn_wmma_f32_16x16x4_f32(false, a, false, b, (short)0, c_hi, false, false);
    frw = (lane < 16) ? c_lo[0] : c_hi[0];   // lane L -> neuron wbase+L
  }
#else
  {
    // Fallback: codegen-confirmed f16 WMMA, same ones-A column-sum trick.
    // Lane l owns column l%16; low lanes feed the lo tile, high lanes the hi tile.
    const int n_self = min(i, nm1);
    float p[8];
    #pragma unroll
    for (int k = 0; k < 8; ++k)
      p[k] = FR[k * N_NEURON + n_self] * W[k * N_NEURON + n_self];
    v16h ones;
    #pragma unroll
    for (int j = 0; j < 16; ++j) ones[j] = (_Float16)1.0f;
    v16h b_lo = {}, b_hi = {};
    if (lane < 16) {
      #pragma unroll
      for (int k = 0; k < 8; ++k) b_lo[k] = (_Float16)p[k];
    } else {
      #pragma unroll
      for (int k = 0; k < 8; ++k) b_hi[k] = (_Float16)p[k];
    }
    v8f c_lo = {}, c_hi = {};
    c_lo = __builtin_amdgcn_wmma_f32_16x16x32_f16(false, ones, false, b_lo, (short)0, c_lo, false, false);
    c_hi = __builtin_amdgcn_wmma_f32_16x16x32_f16(false, ones, false, b_hi, (short)0, c_hi, false, false);
    frw = (lane < 16) ? c_lo[0] : c_hi[0];
  }
#endif

  if (i < N_NEURON) {
    const float inp = input[i];
    const float c   = C[i];
    const float sf  = fmaxf(tanhf(frw), 0.0f);
    const float input_n = fmaxf(inp + c * (sf - inp), 0.0f);  // = max(-C*(inp-sf)+inp, 0)
    int idx = (int)__builtin_roundf(input_n * I_MAX_F);
    idx = idx < 0 ? 0 : (idx > 20 ? 20 : idx);
    frn_out[i]  = lut[idx] * (1.0f / (float)SIM_TIME);
    inpn_out[i] = input_n;
    const int S = (input_n >= FIRE_TH) ? 1 : 0;
    if (__any(S)) {
      if (lane == 0) atomicOr(fired_flag, 1u);
    }
  }
}

// ---------------------------------------------------------------------
// Kernel 3: scalar Fired update + gated W_LatInh update.
// arccos(S) with S in {0,1}:  S=1 -> 0,  S=0 -> pi/2  =>  2*acos(S) in {0, pi}.
__global__ __launch_bounds__(256) void latinh_kernel(
    const float* __restrict__ Fired, const float* __restrict__ W_LatInh,
    const float* __restrict__ inpn,  const unsigned int* __restrict__ fired_flag,
    float* __restrict__ fired_out,   float* __restrict__ wl_out) {
  const int i = blockIdx.x * blockDim.x + threadIdx.x;
  if (i >= N_NEURON) return;
  const float fired = (*fired_flag != 0u) ? 1.0f : 0.0f;
  const float Fnew  = Fired[0] + fired;
  if (i == 0) *fired_out = Fnew;
  const float wl = W_LatInh[i];
  float out;
  if (fired > 0.0f) {
    const float S    = (inpn[i] >= FIRE_TH) ? 1.0f : 0.0f;
    const float term = (S > 0.0f) ? 0.0f : (PI_F * expf(Fnew));
    out = tanhf(wl - term - 1.0f) + 1.0f;
  } else {
    out = wl;
  }
  wl_out[i] = out;
}

// ---------------------------------------------------------------------
extern "C" void kernel_launch(void* const* d_in, const int* in_sizes, int n_in,
                              void* d_out, int out_size, void* d_ws, size_t ws_size,
                              hipStream_t stream) {
  // setup_inputs order: input, FR, C, Fired, W_LatInh, W  (all float32)
  const float* input    = (const float*)d_in[0];
  const float* FR       = (const float*)d_in[1];
  const float* C        = (const float*)d_in[2];
  const float* Fired    = (const float*)d_in[3];
  const float* W_LatInh = (const float*)d_in[4];
  const float* W        = (const float*)d_in[5];

  // outputs concatenated flat: FR_n[N], input_n[N], Fired_new[1], W_LatInh_new[N]
  float* out      = (float*)d_out;
  float* frn      = out;
  float* inpn     = out + N_NEURON;
  float* firednew = out + 2 * N_NEURON;
  float* wlout    = out + 2 * N_NEURON + 1;

  // scratch: 21-float LUT at offset 0, fired flag at offset 128
  float*        lut  = (float*)d_ws;
  unsigned int* flag = (unsigned int*)((char*)d_ws + 128);

  const int blocks = (N_NEURON + 255) / 256;
  izh_lut_kernel<<<1, 32, 0, stream>>>(lut, flag);
  neuron_kernel<<<blocks, 256, 0, stream>>>(input, FR, C, W, lut, flag, frn, inpn);
  latinh_kernel<<<blocks, 256, 0, stream>>>(Fired, W_LatInh, inpn, flag, firednew, wlout);
}
